// GroupQueryAttention_43181601194829
// MI455X (gfx1250) — matrix operand order repaired and checked by reading, not compiled, not run on hardware
//
#include <hip/hip_runtime.h>

// ---------------------------------------------------------------------------
// GQA attention forward for MI455X (gfx1250): wave32, WMMA f16->f32,
// TDM (tensor_load_to_lds) double-buffered GEMM tiles, flash attention.
// ---------------------------------------------------------------------------

typedef _Float16 h8   __attribute__((ext_vector_type(8)));
typedef _Float16 h16  __attribute__((ext_vector_type(16)));
typedef float    f32x8 __attribute__((ext_vector_type(8)));
typedef unsigned int u32x4 __attribute__((ext_vector_type(4)));
typedef int      i32x8 __attribute__((ext_vector_type(8)));
typedef int      i32x4 __attribute__((ext_vector_type(4)));

#define HIDDEN 2048
#define SEQ    2048
#define NHEAD  32
#define DHEAD  64
#define KVHEAD 8
#define QDIM   (NHEAD * DHEAD)    // 2048
#define KVDIM  (KVHEAD * DHEAD)   // 512

__device__ __forceinline__ h16 cat8(h8 lo, h8 hi) {
  return __builtin_shufflevector(lo, hi, 0,1,2,3,4,5,6,7,8,9,10,11,12,13,14,15);
}

__device__ __forceinline__ f32x8 wmma_f16(h16 a, h16 b, f32x8 c) {
  return __builtin_amdgcn_wmma_f32_16x16x32_f16(
      false, a, false, b, (short)0, c, false, false);
}

// ---------------------------------------------------------------------------
// TDM: DMA a [rows x 32] f16 tile (row stride = stride_elems) into LDS.
// D# layout per CDNA5 ISA ch.8: group0 = {count|flags, lds_addr, gaddr, type},
// group1 = {mask/dsize, dims, tile dims, dim0 stride}; groups 2/3 unused (2D).
// ---------------------------------------------------------------------------
__device__ __forceinline__ void tdm_load_tile(unsigned int lds_byte_off,
                                              const _Float16* gsrc,
                                              int rows, int stride_elems) {
  unsigned long long ga = (unsigned long long)(size_t)gsrc;
  u32x4 g0;
  g0[0] = 1u;                                               // count=1, user mode
  g0[1] = lds_byte_off;                                     // LDS dest (bytes)
  g0[2] = (unsigned int)ga;                                 // global_addr[31:0]
  g0[3] = (unsigned int)((ga >> 32) & 0x01FFFFFFull)        // global_addr[56:32]
        | (2u << 30);                                       // type = 2 ("image")
  i32x8 g1;
  g1[0] = 1 << 16;                   // workgroup_mask=0, data_size=1 (2 bytes)
  g1[1] = 32 << 16;                  // tensor_dim0[15:0] = 32
  g1[2] = (rows & 0xFFFF) << 16;     // tensor_dim0 hi=0 | tensor_dim1[15:0]
  g1[3] = 32 << 16;                  // tensor_dim1 hi=0 | tile_dim0 = 32
  g1[4] = rows & 0xFFFF;             // tile_dim1 = rows, tile_dim2 = 0
  g1[5] = stride_elems;              // tensor_dim0_stride[31:0]
  g1[6] = 0;                         // stride0 hi | stride1 lo
  g1[7] = 0;
  i32x4 gz = {0, 0, 0, 0};
#if __clang_major__ >= 23
  i32x8 gz8 = {0, 0, 0, 0, 0, 0, 0, 0};
  __builtin_amdgcn_tensor_load_to_lds(g0, g1, gz, gz, gz8, 0);
#else
  __builtin_amdgcn_tensor_load_to_lds(g0, g1, gz, gz, 0);
#endif
}

// ---------------------------------------------------------------------------
// fp32 -> fp16 conversion (row-major copy)
// ---------------------------------------------------------------------------
__global__ void cvt_f32_to_f16(const float* __restrict__ src,
                               _Float16* __restrict__ dst, int n) {
  int i = blockIdx.x * blockDim.x + threadIdx.x;
  int stride = gridDim.x * blockDim.x;
  for (; i < n; i += stride) dst[i] = (_Float16)src[i];
}

// fp32 [R][C] -> fp16 transposed [C][R], 32x32 LDS tiles
__global__ __launch_bounds__(256)
void cvt_transpose_f16(const float* __restrict__ src, _Float16* __restrict__ dst,
                       int R, int C) {
  __shared__ _Float16 t[32][33];
  const int c0 = blockIdx.x * 32, r0 = blockIdx.y * 32;
  const int tx = threadIdx.x & 31, ty = threadIdx.x >> 5;   // ty = 0..7
#pragma unroll
  for (int i = 0; i < 32; i += 8)
    t[ty + i][tx] = (_Float16)src[(size_t)(r0 + ty + i) * C + c0 + tx];
  __syncthreads();
#pragma unroll
  for (int i = 0; i < 32; i += 8)
    dst[(size_t)(c0 + ty + i) * R + r0 + tx] = t[tx][ty + i];
}

// ---------------------------------------------------------------------------
// WMMA GEMM: C[M,N] = A[M,K](f16,row-major) * Bt[N,K](f16,row-major = B^T)
//            + bias[N](f32)
// Block tile 128x64, 8 waves, wave tile 32x32 (4 WMMA / K-step).
// A/B tiles DMA'd by TDM (wave 0), double-buffered, TENSORcnt handshake.
// out_mode: 0 = f32, 1 = f16, 2 = f16 transposed ([N][M])
// ---------------------------------------------------------------------------
__global__ __launch_bounds__(256, 2)
void gemm_wmma_f16(const _Float16* __restrict__ A, const _Float16* __restrict__ Bt,
                   const float* __restrict__ bias, void* __restrict__ out,
                   int M, int N, int K, int out_mode) {
  __shared__ _Float16 As[2][128][32];   // [buf][m][k]  8KB each
  __shared__ _Float16 Bs[2][64][32];    // [buf][n][k]  4KB each

  const int tid  = threadIdx.x;
  const int lane = tid & 31;
  const int w    = tid >> 5;         // 0..7
  const int wm   = w >> 1;           // 0..3 (32-row strip)
  const int wn   = w & 1;            // 0..1 (32-col strip)
  const int hi   = lane >> 4;
  const int lc   = lane & 15;
  const int m0   = blockIdx.y * 128;
  const int n0   = blockIdx.x * 64;
  const bool issuer = (w == 0);      // wave 0 drives the TDM

  f32x8 c00 = {}, c01 = {}, c10 = {}, c11 = {};

  if (issuer) {
    tdm_load_tile((unsigned int)(size_t)&As[0][0][0], A  + (size_t)m0 * K, 128, K);
    tdm_load_tile((unsigned int)(size_t)&Bs[0][0][0], Bt + (size_t)n0 * K,  64, K);
  }

  const int k0 = hi * 16;   // A-fragment K offset (halves at k0, k0+8)
  const int kb = hi * 16;   // B-fragment K chunk

  for (int k = 0; k < K; k += 32) {
    const int buf = (k >> 5) & 1;
    if (issuer) __builtin_amdgcn_s_wait_tensorcnt(0);   // current tiles landed
    __syncthreads();
    if (issuer && (k + 32 < K)) {                        // prefetch next tiles
      tdm_load_tile((unsigned int)(size_t)&As[buf ^ 1][0][0],
                    A + (size_t)m0 * K + (k + 32), 128, K);
      tdm_load_tile((unsigned int)(size_t)&Bs[buf ^ 1][0][0],
                    Bt + (size_t)n0 * K + (k + 32), 64, K);
    }

    const int ra = wm * 32 + lc;
    h16 a0 = cat8(*(const h8*)&As[buf][ra][k0],      *(const h8*)&As[buf][ra][k0 + 8]);
    h16 a1 = cat8(*(const h8*)&As[buf][ra + 16][k0], *(const h8*)&As[buf][ra + 16][k0 + 8]);
    const int rb = wn * 32 + lc;
    h16 b0 = cat8(*(const h8*)&Bs[buf][rb][kb],      *(const h8*)&Bs[buf][rb][kb + 8]);
    h16 b1 = cat8(*(const h8*)&Bs[buf][rb + 16][kb], *(const h8*)&Bs[buf][rb + 16][kb + 8]);

    c00 = wmma_f16(a0, b0, c00);
    c01 = wmma_f16(a0, b1, c01);
    c10 = wmma_f16(a1, b0, c10);
    c11 = wmma_f16(a1, b1, c11);
    __syncthreads();
  }

#pragma unroll
  for (int v = 0; v < 8; ++v) {
    const int r0 = m0 + wm * 32 + v + 8 * hi;
    const int r1 = r0 + 16;
    const int col0 = n0 + wn * 32 + lc;
    const int col1 = col0 + 16;
    const float v00 = c00[v] + bias[col0];
    const float v01 = c01[v] + bias[col1];
    const float v10 = c10[v] + bias[col0];
    const float v11 = c11[v] + bias[col1];
    if (out_mode == 0) {
      float* o = (float*)out;
      o[(size_t)r0 * N + col0] = v00; o[(size_t)r0 * N + col1] = v01;
      o[(size_t)r1 * N + col0] = v10; o[(size_t)r1 * N + col1] = v11;
    } else if (out_mode == 1) {
      _Float16* o = (_Float16*)out;
      o[(size_t)r0 * N + col0] = (_Float16)v00; o[(size_t)r0 * N + col1] = (_Float16)v01;
      o[(size_t)r1 * N + col0] = (_Float16)v10; o[(size_t)r1 * N + col1] = (_Float16)v11;
    } else {                                   // transposed f16 output [N][M]
      _Float16* o = (_Float16*)out;
      o[(size_t)col0 * M + r0] = (_Float16)v00; o[(size_t)col1 * M + r0] = (_Float16)v01;
      o[(size_t)col0 * M + r1] = (_Float16)v10; o[(size_t)col1 * M + r1] = (_Float16)v11;
    }
  }
}

// ---------------------------------------------------------------------------
// PV step with transposed V (Vt: [KVDIM][SEQ]) -> contiguous 16B loads.
// ---------------------------------------------------------------------------
__device__ __forceinline__ f32x8 pv_step(const _Float16* __restrict__ Vt,
                                         int j, int kb, int drow,
                                         h16 pa, f32x8 o) {
  const _Float16* p = Vt + (size_t)drow * SEQ + j + kb;
  h16 bv = cat8(*(const h8*)p, *(const h8*)(p + 8));
  return wmma_f16(pa, bv, o);
}

// ---------------------------------------------------------------------------
// Flash attention: one wave per (head, 16-query-row block); causal analytic;
// online softmax; P converted C-layout -> A-layout via 1KB LDS tile.
// ---------------------------------------------------------------------------
__global__ __launch_bounds__(32)
void attn_wmma(const _Float16* __restrict__ Q, const _Float16* __restrict__ Km,
               const _Float16* __restrict__ Vt, _Float16* __restrict__ O) {
  __shared__ _Float16 plds[16][32];

  const int lane  = threadIdx.x;
  const int hi    = lane >> 4;
  const int lc    = lane & 15;
  const int qb    = blockIdx.x;
  const int h     = blockIdx.y;
  const int kvh   = h >> 2;             // repeat_interleave(4)
  const int qbase = qb * 16;
  const float scale = 0.125f;           // 1/sqrt(64)

  h16 aq[2];
  {
    const _Float16* qrow = Q + (size_t)(qbase + lc) * QDIM + h * DHEAD;
    const int k0 = hi * 16;
#pragma unroll
    for (int t = 0; t < 2; ++t)
      aq[t] = cat8(*(const h8*)&qrow[t * 32 + k0],
                   *(const h8*)&qrow[t * 32 + k0 + 8]);
  }

  f32x8 o0 = {}, o1 = {}, o2 = {}, o3 = {};
  float mrow[8], lrow[8];
#pragma unroll
  for (int v = 0; v < 8; ++v) { mrow[v] = -3.0e38f; lrow[v] = 0.0f; }

  const int qmax = qbase + 15;
  for (int j = 0; j <= qmax; j += 32) {
    float p[2][8];

    // ---- S = Q K^T, scaled + causal mask ----
#pragma unroll
    for (int stile = 0; stile < 2; ++stile) {
      f32x8 s = {};
      const int key = j + stile * 16 + lc;
      const _Float16* krow = Km + (size_t)key * KVDIM + kvh * DHEAD;
      const int kb = hi * 16;
#pragma unroll
      for (int t = 0; t < 2; ++t) {
        h16 bk = cat8(*(const h8*)&krow[t * 32 + kb],
                      *(const h8*)&krow[t * 32 + kb + 8]);
        s = wmma_f16(aq[t], bk, s);
      }
#pragma unroll
      for (int v = 0; v < 8; ++v) {
        const int qr = qbase + v + 8 * hi;
        float val = s[v] * scale;
        if (key > qr) val = -1.0e30f;
        p[stile][v] = val;
      }
    }

    // ---- online softmax (16-lane half-wave row reductions) ----
#pragma unroll
    for (int v = 0; v < 8; ++v) {
      float mx = fmaxf(p[0][v], p[1][v]);
#pragma unroll
      for (int off = 8; off >= 1; off >>= 1)
        mx = fmaxf(mx, __shfl_xor(mx, off, 32));
      const float mnew  = fmaxf(mrow[v], mx);
      const float alpha = __expf(mrow[v] - mnew);
      const float e0 = __expf(p[0][v] - mnew);
      const float e1 = __expf(p[1][v] - mnew);
      float rs = e0 + e1;
#pragma unroll
      for (int off = 8; off >= 1; off >>= 1)
        rs += __shfl_xor(rs, off, 32);
      lrow[v] = lrow[v] * alpha + rs;
      mrow[v] = mnew;
      o0[v] *= alpha; o1[v] *= alpha; o2[v] *= alpha; o3[v] *= alpha;
      p[0][v] = e0; p[1][v] = e1;
    }

    // ---- P: C-layout -> LDS -> A-layout fragment ----
#pragma unroll
    for (int v = 0; v < 8; ++v) {
      const int r = v + 8 * hi;
      plds[r][lc]      = (_Float16)p[0][v];
      plds[r][lc + 16] = (_Float16)p[1][v];
    }
    __syncthreads();
    h16 pa;
    {
      const int k0 = hi * 16;
      pa = cat8(*(const h8*)&plds[lc][k0], *(const h8*)&plds[lc][k0 + 8]);
    }
    __syncthreads();

    // ---- O += P * V (Vt rows are d, contiguous over keys) ----
    const int kb = hi * 16;
    const int db = kvh * DHEAD + lc;
    o0 = pv_step(Vt, j, kb, db +  0, pa, o0);
    o1 = pv_step(Vt, j, kb, db + 16, pa, o1);
    o2 = pv_step(Vt, j, kb, db + 32, pa, o2);
    o3 = pv_step(Vt, j, kb, db + 48, pa, o3);
  }

#pragma unroll
  for (int v = 0; v < 8; ++v) {
    const float inv = 1.0f / lrow[v];
    const int qr = qbase + v + 8 * hi;
    _Float16* orow = O + (size_t)qr * QDIM + h * DHEAD;
    orow[ 0 + lc] = (_Float16)(o0[v] * inv);
    orow[16 + lc] = (_Float16)(o1[v] * inv);
    orow[32 + lc] = (_Float16)(o2[v] * inv);
    orow[48 + lc] = (_Float16)(o3[v] * inv);
  }
}

// ---------------------------------------------------------------------------
// Host launcher
// ---------------------------------------------------------------------------
extern "C" void kernel_launch(void* const* d_in, const int* in_sizes, int n_in,
                              void* d_out, int out_size, void* d_ws, size_t ws_size,
                              hipStream_t stream) {
  (void)in_sizes; (void)n_in; (void)out_size; (void)ws_size;
  const float* x  = (const float*)d_in[0];
  // d_in[1] = causal_mask (mask applied analytically)
  const float* Wq = (const float*)d_in[2];
  const float* bq = (const float*)d_in[3];
  const float* Wk = (const float*)d_in[4];
  const float* bk = (const float*)d_in[5];
  const float* Wv = (const float*)d_in[6];
  const float* bv = (const float*)d_in[7];
  const float* Wo = (const float*)d_in[8];
  const float* bo = (const float*)d_in[9];
  float* out = (float*)d_out;

  char* ws = (char*)d_ws;
  const size_t MB = 1024ull * 1024ull;
  _Float16* xh  = (_Float16*)(ws +  0 * MB);   // [2048][2048]       8 MB
  _Float16* WqT = (_Float16*)(ws +  8 * MB);   // [2048][2048] = Wq^T 8 MB
  _Float16* WkT = (_Float16*)(ws + 16 * MB);   // [512][2048]  = Wk^T 2 MB
  _Float16* WvT = (_Float16*)(ws + 18 * MB);   // [512][2048]  = Wv^T 2 MB
  _Float16* WoT = (_Float16*)(ws + 20 * MB);   // [2048][2048] = Wo^T 8 MB
  _Float16* Qh  = (_Float16*)(ws + 28 * MB);   // [2048][2048]       8 MB
  _Float16* Kh  = (_Float16*)(ws + 36 * MB);   // [2048][512]        2 MB
  _Float16* Vt  = (_Float16*)(ws + 38 * MB);   // [512][2048] = V^T  2 MB
  _Float16* Oh  = (_Float16*)(ws + 40 * MB);   // [2048][2048]       8 MB (48 total)

  // staging: x row-major f16; weights transposed f16 (so GEMM B-frags are
  // contiguous and TDM-tileable)
  cvt_f32_to_f16<<<512, 256, 0, stream>>>(x, xh, SEQ * HIDDEN);
  cvt_transpose_f16<<<dim3(QDIM  / 32, HIDDEN / 32), 256, 0, stream>>>(Wq, WqT, HIDDEN, QDIM);
  cvt_transpose_f16<<<dim3(KVDIM / 32, HIDDEN / 32), 256, 0, stream>>>(Wk, WkT, HIDDEN, KVDIM);
  cvt_transpose_f16<<<dim3(KVDIM / 32, HIDDEN / 32), 256, 0, stream>>>(Wv, WvT, HIDDEN, KVDIM);
  cvt_transpose_f16<<<dim3(HIDDEN / 32, QDIM / 32), 256, 0, stream>>>(Wo, WoT, QDIM, HIDDEN);

  // projections: Q,K row-major f16; V written transposed for the PV WMMA
  gemm_wmma_f16<<<dim3(QDIM  / 64, SEQ / 128), 256, 0, stream>>>(
      xh, WqT, bq, (void*)Qh, SEQ, QDIM,  HIDDEN, 1);
  gemm_wmma_f16<<<dim3(KVDIM / 64, SEQ / 128), 256, 0, stream>>>(
      xh, WkT, bk, (void*)Kh, SEQ, KVDIM, HIDDEN, 1);
  gemm_wmma_f16<<<dim3(KVDIM / 64, SEQ / 128), 256, 0, stream>>>(
      xh, WvT, bv, (void*)Vt, SEQ, KVDIM, HIDDEN, 2);

  // causal GQA flash attention
  attn_wmma<<<dim3(SEQ / 16, NHEAD), 32, 0, stream>>>(Qh, Kh, Vt, Oh);

  // output projection (f32 + bias)
  gemm_wmma_f16<<<dim3(HIDDEN / 64, SEQ / 128), 256, 0, stream>>>(
      Oh, WoT, bo, (void*)out, SEQ, HIDDEN, QDIM, 0);
}
